// GAT_Module_90623809945643
// MI455X (gfx1250) — compile-verified, matrix operand order checked by reading, not run on hardware
//
#include <hip/hip_runtime.h>
#include <cstdint>
#include <cstddef>

// ---------------------------------------------------------------------------
// GAT (2-layer) for MI455X / gfx1250.
// N=50000 nodes, E=800000 edges, H=4 heads, D=64.
// GEMM on the WMMA pipe (bf16 in, f32 accumulate), with W staged into LDS by
// the Tensor Data Mover (tensor_load_to_lds + s_wait_tensorcnt). Edge passes
// are the memory/atomic-bound bottleneck: coalesced float2 access, L2-resident
// f32 atomics, and global_prefetch lookahead.
// ---------------------------------------------------------------------------

#define GAT_N  50000
#define GAT_E  800000
#define GAT_H  4
#define GAT_D  64
#define GAT_HD 256   // H*D
#define NEG_SLOPE 0.2f
#define WREP_ELEMS (GAT_D * GAT_HD)   // 16384 bf16 = 32 KB

typedef __attribute__((ext_vector_type(16))) __bf16 v16bf;
typedef __attribute__((ext_vector_type(8)))  float  v8f;
typedef unsigned int u32x4 __attribute__((ext_vector_type(4)));
typedef int          i32x4 __attribute__((ext_vector_type(4)));
typedef int          i32x8 __attribute__((ext_vector_type(8)));

// -------------------------------- helpers ----------------------------------

__device__ __forceinline__ void atomicMaxFloat(float* addr, float val) {
  // IEEE trick: non-negative floats compare as ints, negative floats compare
  // reversed as uints. Works with init = -inf.
  if (!(val < 0.0f)) {
    atomicMax((int*)addr, __float_as_int(val));
  } else {
    atomicMin((unsigned int*)addr, __float_as_uint(val));
  }
}

// Issue a TDM 1-D copy of `nbytes/2` bf16 elements from global `gaddr` to LDS
// offset `lds_off`, per CDNA5 ISA D# layout (group0/group1; groups 2-3 zero).
__device__ __forceinline__ void tdm_load_1d_bf16(unsigned lds_off,
                                                 unsigned long long gaddr,
                                                 unsigned nelem) {
  u32x4 g0;
  g0[0] = 1u;                                   // count=1, is_restore=0
  g0[1] = lds_off;                              // lds_addr
  g0[2] = (unsigned)(gaddr & 0xFFFFFFFFull);    // global_addr[31:0]
  g0[3] = (unsigned)((gaddr >> 32) & 0x01FFFFFFull) | (2u << 30); // addr[56:32], type=2

  i32x8 g1;
  g1[0] = (int)(1u << 16);                      // workgroup_mask=0, data_size=1 (2B)
  g1[1] = (int)((nelem & 0xFFFFu) << 16);       // tensor_dim0[15:0]
  g1[2] = (int)(((nelem >> 16) & 0xFFFFu) | (1u << 16)); // tensor_dim0[31:16], tensor_dim1=1
  g1[3] = (int)((nelem & 0xFFFFu) << 16);       // tile_dim0 = nelem
  g1[4] = 0;                                    // tile_dim1=0 (unused), tile_dim2=0
  g1[5] = (int)nelem;                           // tensor_dim0_stride[31:0]
  g1[6] = 0;                                    // stride hi, dim1_stride lo
  g1[7] = 0;

  i32x4 g2 = {};
  i32x4 g3 = {};
#if __clang_major__ >= 23
  i32x8 g4 = {};
  __builtin_amdgcn_tensor_load_to_lds(g0, g1, g2, g3, g4, 0);
#else
  __builtin_amdgcn_tensor_load_to_lds(g0, g1, g2, g3, 0);
#endif
  __builtin_amdgcn_s_wait_tensorcnt(0);
}

// --------------------------- elementwise kernels ---------------------------

__global__ void cast_f32_to_bf16(const float* __restrict__ in,
                                 __bf16* __restrict__ out, int n) {
  int i = blockIdx.x * blockDim.x + threadIdx.x;
  if (i < n) out[i] = (__bf16)in[i];
}

__global__ void init_mz(float* __restrict__ m, float* __restrict__ z, int n) {
  int i = blockIdx.x * blockDim.x + threadIdx.x;
  if (i < n) { m[i] = -__builtin_inff(); z[i] = 0.0f; }
}

__global__ void zero_f32(float* __restrict__ p, int n) {
  int i = blockIdx.x * blockDim.x + threadIdx.x;
  if (i < n) p[i] = 0.0f;
}

// ------------------------- W repack (fragment-ready) -----------------------
// Brep[((tn*2+kt)*32 + lane)*16 + i] = bf16(W[k, col]) where the 16 values a
// lane feeds into the WMMA B operand are contiguous (one 32B chunk per lane):
//   l15 = lane&15, half = lane>>4, kbase = kt*32 + half*8
//   k   = i<8 ? kbase+i : kbase+16+(i-8),  col = tn*16 + l15
__global__ void repack_W_bf16(const float* __restrict__ W,
                              __bf16* __restrict__ Brep) {
  int idx = blockIdx.x * blockDim.x + threadIdx.x;
  if (idx >= WREP_ELEMS) return;
  int i    = idx & 15;
  int lane = (idx >> 4) & 31;
  int kt   = (idx >> 9) & 1;
  int tn   = idx >> 10;                        // 0..15
  int l15  = lane & 15, half = lane >> 4;
  int kbase = kt * 32 + half * 8;
  int k    = (i < 8) ? (kbase + i) : (kbase + 16 + (i - 8));
  int col  = tn * 16 + l15;
  Brep[idx] = (__bf16)W[k * GAT_HD + col];
}

// ------------------------------ WMMA GEMM ----------------------------------
// C[M,256] (f32) = A[M,64] (bf16) x B[64,256] (bf16; pre-repacked Brep).
// One wave32 per 16x16 output tile, K=64 -> two v_wmma_f32_16x16x32_bf16.
// Wave 0 of each block DMAs the full 32KB Brep into LDS via the TDM; waves
// then read B fragments with two ds_load_b128 per K-step.
__global__ void wmma_gemm_bf16(const __bf16* __restrict__ A,
                               const __bf16* __restrict__ Brep,
                               float* __restrict__ C, int M) {
  __shared__ __align__(128) __bf16 Blds[WREP_ELEMS];

  if (threadIdx.x < 32) {  // one wave issues the tensor DMA
    tdm_load_1d_bf16((unsigned)(uintptr_t)&Blds[0],
                     (unsigned long long)(uintptr_t)Brep, WREP_ELEMS);
  }
  __syncthreads();

  const int lane   = threadIdx.x & 31;
  const int wave   = threadIdx.x >> 5;
  const int wpb    = blockDim.x >> 5;
  const int tilesN = GAT_HD / 16;              // 16
  int tile = blockIdx.x * wpb + wave;
  int numTiles = (M / 16) * tilesN;
  if (tile >= numTiles) return;

  const int tm   = tile / tilesN;
  const int tn   = tile % tilesN;
  const int l15  = lane & 15;
  const int half = lane >> 4;                  // 0 or 1
  const int m    = tm * 16 + l15;

  v8f acc = {};
  const __bf16* arow = A + (size_t)m * GAT_D;

#pragma unroll
  for (int kt = 0; kt < 2; ++kt) {             // K blocks of 32
    const int kbase = kt * 32 + half * 8;
    union { uint4 u[2]; v16bf v; } au, bu;
    au.u[0] = *(const uint4*)(arow + kbase);        // K = kbase..kbase+7
    au.u[1] = *(const uint4*)(arow + kbase + 16);   // K = kbase+16..kbase+23
    const uint4* bl =
        (const uint4*)(Blds + ((size_t)(tn * 2 + kt) * 32 + lane) * 16);
    bu.u[0] = bl[0];
    bu.u[1] = bl[1];
    acc = __builtin_amdgcn_wmma_f32_16x16x32_bf16(
        /*neg_a=*/false, au.v, /*neg_b=*/false, bu.v,
        /*c_mod=*/(short)0, acc, /*reuse_a=*/false, /*reuse_b=*/false);
  }

  const int row0 = tm * 16 + half * 8;         // C/D: VGPR v -> M=v+8*half
  const int col  = tn * 16 + l15;
#pragma unroll
  for (int v = 0; v < 8; ++v)
    C[(size_t)(row0 + v) * GAT_HD + col] = acc[v];
}

// --------------------------- attention scalars -----------------------------
__global__ void attn_proj(const float* __restrict__ h,
                          const float* __restrict__ al,
                          const float* __restrict__ ar,
                          float* __restrict__ el, float* __restrict__ er,
                          int nNH) {
  int i = blockIdx.x * blockDim.x + threadIdx.x;
  if (i >= nNH) return;
  int node = i >> 2, hh = i & 3;
  const float* hp = h + (size_t)node * GAT_HD + hh * GAT_D;
  const float* lp = al + hh * GAT_D;
  const float* rp = ar + hh * GAT_D;
  float sl = 0.f, sr = 0.f;
#pragma unroll 8
  for (int d = 0; d < GAT_D; ++d) {
    float v = hp[d];
    sl = fmaf(v, lp[d], sl);
    sr = fmaf(v, rp[d], sr);
  }
  el[i] = sl; er[i] = sr;
}

// ------------------------------ edge passes --------------------------------

__global__ void edge_attn_max(const int* __restrict__ src,
                              const int* __restrict__ dst,
                              const float* __restrict__ el,
                              const float* __restrict__ er,
                              float* __restrict__ ea, float* __restrict__ m,
                              int EH) {
  int i = blockIdx.x * blockDim.x + threadIdx.x;
  if (i >= EH) return;
  int ed = i >> 2, hh = i & 3;
  int s = src[ed], dnode = dst[ed];
  float v = el[s * GAT_H + hh] + er[dnode * GAT_H + hh];
  v = (v > 0.f) ? v : v * NEG_SLOPE;
  ea[i] = v;
  atomicMaxFloat(&m[dnode * GAT_H + hh], v);
}

__global__ void edge_softmax_num(const int* __restrict__ dst,
                                 const float* __restrict__ m,
                                 float* __restrict__ ea, float* __restrict__ z,
                                 int EH) {
  int i = blockIdx.x * blockDim.x + threadIdx.x;
  if (i >= EH) return;
  int ed = i >> 2, hh = i & 3;
  int dnode = dst[ed];
  float a = __expf(ea[i] - m[dnode * GAT_H + hh]);
  ea[i] = a;
  atomicAdd(&z[dnode * GAT_H + hh], a);
}

// heavy pass: agg[dst,h,:] += h[src,h,:] * alpha. One wave32 per (edge,head);
// each lane moves a float2. Grid-stride so each wave prefetches its next
// source row (global_prefetch_b8).
__global__ void edge_aggregate(const int* __restrict__ src,
                               const int* __restrict__ dst,
                               const float* __restrict__ ea,
                               const float* __restrict__ z,
                               const float* __restrict__ h,
                               float* __restrict__ agg, int EH) {
  const int lane   = threadIdx.x & 31;
  const int wave   = (blockIdx.x * blockDim.x + threadIdx.x) >> 5;
  const int nwaves = (gridDim.x * blockDim.x) >> 5;
  for (int i = wave; i < EH; i += nwaves) {
    int ed = i >> 2, hh = i & 3;
    int s = src[ed], dnode = dst[ed];
    float alpha = ea[i] / z[dnode * GAT_H + hh];

    int ni = i + nwaves;
    if (ni < EH) {
      int ns = src[ni >> 2];
      __builtin_prefetch(h + (size_t)ns * GAT_HD + (ni & 3) * GAT_D + lane * 2,
                         0, 0);
    }

    const float2* hp = (const float2*)(h + (size_t)s * GAT_HD + hh * GAT_D);
    float*        op = agg + (size_t)dnode * GAT_HD + hh * GAT_D + lane * 2;
    float2 v = hp[lane];
    atomicAdd(op + 0, v.x * alpha);
    atomicAdd(op + 1, v.y * alpha);
  }
}

__global__ void finalize_layer(const float* __restrict__ agg,
                               const float* __restrict__ bias,
                               float* __restrict__ out, int nND,
                               int apply_tanh) {
  int i = blockIdx.x * blockDim.x + threadIdx.x;
  if (i >= nND) return;
  int node = i >> 6, d = i & 63;
  float acc = 0.f;
#pragma unroll
  for (int hh = 0; hh < GAT_H; ++hh) {
    float t = agg[(size_t)node * GAT_HD + hh * GAT_D + d] + bias[hh * GAT_D + d];
    if (apply_tanh) t = tanhf(t);
    acc += t;
  }
  out[i] = acc * 0.25f;
}

// ------------------------------ host driver --------------------------------

static void run_gat_layer(const float* xin, const int* src, const int* dst,
                          const float* W, const float* al, const float* ar,
                          const float* bias, float* out, int apply_tanh,
                          __bf16* Xb, __bf16* Brep, float* h, float* agg,
                          float* el, float* er, float* m, float* z, float* ea,
                          hipStream_t stream) {
  const int TB  = 256;                                   // 8 waves (wave32)
  const int ND  = GAT_N * GAT_D;                         // 3.2M
  const int NH  = GAT_N * GAT_H;                         // 200K
  const int NHD = GAT_N * GAT_HD;                        // 12.8M
  const int EH  = GAT_E * GAT_H;                         // 3.2M

  cast_f32_to_bf16<<<(ND + TB - 1) / TB, TB, 0, stream>>>(xin, Xb, ND);
  repack_W_bf16<<<(WREP_ELEMS + TB - 1) / TB, TB, 0, stream>>>(W, Brep);

  // (N/16)*(256/16) = 50000 tiles, 8 tiles (waves) per 256-thread block
  int numTiles = (GAT_N / 16) * (GAT_HD / 16);
  wmma_gemm_bf16<<<(numTiles + 7) / 8, TB, 0, stream>>>(Xb, Brep, h, GAT_N);

  attn_proj<<<(NH + TB - 1) / TB, TB, 0, stream>>>(h, al, ar, el, er, NH);
  init_mz<<<(NH + TB - 1) / TB, TB, 0, stream>>>(m, z, NH);
  zero_f32<<<(NHD + TB - 1) / TB, TB, 0, stream>>>(agg, NHD);

  edge_attn_max<<<(EH + TB - 1) / TB, TB, 0, stream>>>(src, dst, el, er, ea, m, EH);
  edge_softmax_num<<<(EH + TB - 1) / TB, TB, 0, stream>>>(dst, m, ea, z, EH);
  edge_aggregate<<<4096, TB, 0, stream>>>(src, dst, ea, z, h, agg, EH);

  finalize_layer<<<(ND + TB - 1) / TB, TB, 0, stream>>>(agg, bias, out, ND,
                                                        apply_tanh);
}

extern "C" void kernel_launch(void* const* d_in, const int* in_sizes, int n_in,
                              void* d_out, int out_size, void* d_ws,
                              size_t ws_size, hipStream_t stream) {
  (void)in_sizes; (void)n_in; (void)out_size; (void)ws_size;

  const float* x   = (const float*)d_in[0];
  const int*   src = (const int*)d_in[1];
  const int*   dst = (const int*)d_in[2];
  const float* W1  = (const float*)d_in[3];
  const float* al1 = (const float*)d_in[4];
  const float* ar1 = (const float*)d_in[5];
  const float* b1  = (const float*)d_in[6];
  const float* W2  = (const float*)d_in[7];
  const float* al2 = (const float*)d_in[8];
  const float* ar2 = (const float*)d_in[9];
  const float* b2  = (const float*)d_in[10];
  float* out = (float*)d_out;

  // ---- carve workspace (256B aligned chunks) ----
  size_t off = 0;
  auto carve = [&](size_t bytes) -> void* {
    void* p = (char*)d_ws + off;
    off += (bytes + 255) & ~(size_t)255;
    return p;
  };
  float*  h    = (float*)carve((size_t)GAT_N * GAT_HD * 4);  // 51.2 MB
  float*  agg  = (float*)carve((size_t)GAT_N * GAT_HD * 4);  // 51.2 MB
  float*  ea   = (float*)carve((size_t)GAT_E * GAT_H * 4);   // 12.8 MB
  float*  y    = (float*)carve((size_t)GAT_N * GAT_D * 4);   // 12.8 MB
  float*  el   = (float*)carve((size_t)GAT_N * GAT_H * 4);
  float*  er   = (float*)carve((size_t)GAT_N * GAT_H * 4);
  float*  m    = (float*)carve((size_t)GAT_N * GAT_H * 4);
  float*  z    = (float*)carve((size_t)GAT_N * GAT_H * 4);
  __bf16* Xb   = (__bf16*)carve((size_t)GAT_N * GAT_D * 2);  // 6.4 MB
  __bf16* Brep = (__bf16*)carve((size_t)WREP_ELEMS * 2);     // 32 KB

  // layer 1: tanh activation, result -> y
  run_gat_layer(x, src, dst, W1, al1, ar1, b1, y, /*tanh=*/1,
                Xb, Brep, h, agg, el, er, m, z, ea, stream);
  // layer 2: no activation, result -> d_out
  run_gat_layer(y, src, dst, W2, al2, ar2, b2, out, /*tanh=*/0,
                Xb, Brep, h, agg, el, er, m, z, ea, stream);
}
